// AvaAttention_40054865002615
// MI455X (gfx1250) — matrix operand order repaired and checked by reading, not compiled, not run on hardware
//
#include <hip/hip_runtime.h>

// ---------------------------------------------------------------------------
// AvaAttention for MI455X (gfx1250): bf16 WMMA GEMMs + fused-RoPE flash attn.
// - LDS tiles stored pre-swizzled into WMMA fragment layout (32B/lane blocks,
//   read with two ds_load_b128); staging stores widened to b32/b64.
// - GEMM: 64x256 block tile, 8 waves (2Mx4N), 8 WMMAs/wave/K-step, double-
//   buffered LDS software pipeline (one barrier per K-step).
// - Flash attention: block = 8 waves sharing (b,h); cooperative fragment-
//   swizzled K(RoPE)/V staging; online softmax via half-wave shuffles
//   matching the CDNA5 C-layout.
// ---------------------------------------------------------------------------

typedef __attribute__((ext_vector_type(16))) __bf16 v16bf;
typedef __attribute__((ext_vector_type(8)))  __bf16 v8bf;
typedef __attribute__((ext_vector_type(4)))  __bf16 v4bf;
typedef __attribute__((ext_vector_type(2)))  __bf16 v2bf;
typedef __attribute__((ext_vector_type(8)))  float  v8f;
typedef __attribute__((ext_vector_type(4)))  float  v4f;

#define B_   2
#define S_   2048
#define HID_ 2048
#define NH_  32
#define KVH_ 8
#define HD_  64

__device__ __forceinline__ v8f wmma_bf16(v16bf a, v16bf b, v8f c) {
  return __builtin_amdgcn_wmma_f32_16x16x32_bf16(false, a, false, b,
                                                 (short)0, c, false, false);
}

__device__ __forceinline__ v16bf cat16(v8bf lo, v8bf hi) {
  return __builtin_shufflevector(lo, hi, 0, 1, 2, 3, 4, 5, 6, 7, 8, 9, 10, 11,
                                 12, 13, 14, 15);
}
__device__ __forceinline__ v8bf cat8(v4bf lo, v4bf hi) {
  return __builtin_shufflevector(lo, hi, 0, 1, 2, 3, 4, 5, 6, 7);
}

__device__ __forceinline__ void lds_wait_wave() {
#if __has_builtin(__builtin_amdgcn_s_wait_dscnt)
  __builtin_amdgcn_s_wait_dscnt(0);
#endif
}

// ---------------------------------------------------------------------------
// GEMM: C[M,N] = A[M,K] * W[K,N], fp32 src/dst, bf16 WMMA compute.
// Block 256 threads = 8 waves as 2(M)x4(N); tile 64(M) x 256(N), K step 32.
// Wave computes 32x64 = 2 A-frags x 4 B-frags = 8 WMMAs per K-step.
// B staging loads row-pairs so swizzled stores are packed b32.
// ---------------------------------------------------------------------------
#define TM 64
#define TN 256
#define TK 32

__global__ __launch_bounds__(256) void gemm_bf16_wmma_kernel(
    const float* __restrict__ A, const float* __restrict__ W,
    float* __restrict__ C, int M, int K, int N) {
  __shared__ alignas(16) __bf16 As[2][TM][TK + 8];    // 2 x 5KB
  __shared__ alignas(16) __bf16 BsF[2][16][32][16];   // 2 x 16KB, frag-swizzled

  const int tid  = threadIdx.x;
  const int lane = tid & 31;
  const int wave = tid >> 5;
  const int mw   = wave & 1;    // 2 M-strips of 32 rows
  const int nw   = wave >> 1;   // 4 N-strips of 64 cols
  const int bm   = blockIdx.y * TM;
  const int bn   = blockIdx.x * TN;

  const int cn    = lane & 15;
  const int akb   = (lane >> 4) << 4;   // A-layout K base (0/16)
  const int arow0 = mw * 32 + cn;
  const int arow1 = arow0 + 16;

  v8f acc[2][4];
#pragma unroll
  for (int mi = 0; mi < 2; ++mi)
#pragma unroll
    for (int c4 = 0; c4 < 4; ++c4) acc[mi][c4] = v8f{};

  auto load_tile = [&](int k0, v4f (&ax)[2], v4f (&bx0)[4], v4f (&bx1)[4]) {
#pragma unroll
    for (int t = 0; t < 2; ++t) {
      int li = tid + t * 256;           // 0..511 quads of A (64x32)
      int m  = li >> 3, kk = (li & 7) * 4;
      ax[t] = *(const v4f*)&A[(size_t)(bm + m) * K + (k0 + kk)];
    }
#pragma unroll
    for (int t = 0; t < 4; ++t) {
      int li  = tid + t * 256;          // 0..1023 row-pair quads of B (32x256)
      int kk  = (li >> 6) * 2;          // even row of the pair
      int n   = (li & 63) * 4;
      bx0[t] = *(const v4f*)&W[(size_t)(k0 + kk) * N + (bn + n)];
      bx1[t] = *(const v4f*)&W[(size_t)(k0 + kk + 1) * N + (bn + n)];
    }
  };
  auto store_tile = [&](int buf, const v4f (&ax)[2], const v4f (&bx0)[4],
                        const v4f (&bx1)[4]) {
#pragma unroll
    for (int t = 0; t < 2; ++t) {
      int li = tid + t * 256;
      int m  = li >> 3, kk = (li & 7) * 4;
      v4bf y;
      y[0] = (__bf16)ax[t][0]; y[1] = (__bf16)ax[t][1];
      y[2] = (__bf16)ax[t][2]; y[3] = (__bf16)ax[t][3];
      *(v4bf*)&As[buf][m][kk] = y;
    }
#pragma unroll
    for (int t = 0; t < 4; ++t) {
      int li = tid + t * 256;
      int kk = (li >> 6) * 2;
      int n  = (li & 63) * 4;
#pragma unroll
      for (int j = 0; j < 4; ++j) {
        int nn = n + j;
        v2bf y;                         // (kk, kk+1) adjacent in lane block
        y[0] = (__bf16)bx0[t][j];
        y[1] = (__bf16)bx1[t][j];
        *(v2bf*)&BsF[buf][nn >> 4][(kk & 16) + (nn & 15)][kk & 15] = y;
      }
    }
  };

  // ---- pipeline prologue ----
  v4f ax[2], bx0[4], bx1[4];
  load_tile(0, ax, bx0, bx1);
  store_tile(0, ax, bx0, bx1);
  __syncthreads();

  for (int k0 = 0; k0 < K; k0 += TK) {
    const int buf   = (k0 / TK) & 1;
    const bool more = (k0 + TK) < K;
    v4f axn[2], bxn0[4], bxn1[4];
    if (more) load_tile(k0 + TK, axn, bxn0, bxn1);  // overlap with WMMAs

    // ---- fragments (all loads issued, then WMMA burst) ----
    v16bf af0 = cat16(*(const v8bf*)&As[buf][arow0][akb],
                      *(const v8bf*)&As[buf][arow0][akb + 8]);
    v16bf af1 = cat16(*(const v8bf*)&As[buf][arow1][akb],
                      *(const v8bf*)&As[buf][arow1][akb + 8]);
    v16bf bfr[4];
#pragma unroll
    for (int c4 = 0; c4 < 4; ++c4) {
      const __bf16* p = &BsF[buf][nw * 4 + c4][lane][0];
      bfr[c4] = cat16(*(const v8bf*)p, *(const v8bf*)(p + 8));
    }
#pragma unroll
    for (int c4 = 0; c4 < 4; ++c4) {
      acc[0][c4] = wmma_bf16(af0, bfr[c4], acc[0][c4]);
      acc[1][c4] = wmma_bf16(af1, bfr[c4], acc[1][c4]);
    }

    if (more) store_tile(buf ^ 1, axn, bxn0, bxn1);  // other buffer: no race
    __syncthreads();
  }

  // ---- epilogue: C layout (lanes 0-15 rows 0-7, lanes 16-31 rows 8-15) ----
  const int rbase = (lane >> 4) << 3;
#pragma unroll
  for (int mi = 0; mi < 2; ++mi) {
#pragma unroll
    for (int c4 = 0; c4 < 4; ++c4) {
#pragma unroll
      for (int v = 0; v < 8; ++v) {
        int grow = bm + mw * 32 + mi * 16 + rbase + v;
        int gcol = bn + nw * 64 + c4 * 16 + cn;
        C[(size_t)grow * N + gcol] = acc[mi][c4][v];
      }
    }
  }
}

// ---------------------------------------------------------------------------
// Flash attention, non-causal, GQA groups=4, head-indexed RoPE fused.
// ---------------------------------------------------------------------------
__global__ __launch_bounds__(256) void flash_attn_kernel(
    const float* __restrict__ Qp, const float* __restrict__ Kp,
    const float* __restrict__ Vp, const float* __restrict__ cosT,
    const float* __restrict__ sinT, float* __restrict__ Ao) {
  __shared__ alignas(16) __bf16 Kl[2][2][32][16];  // [keychunk][dhalf][lane][i]
  __shared__ alignas(16) __bf16 Vl[4][32][16];     // [dchunk][lane][i]
  __shared__ alignas(16) __bf16 Plds[8][16][40];   // per-wave P tile

  const int bh   = blockIdx.x >> 4;   // b*NH + h
  const int qblk = blockIdx.x & 15;
  const int b    = bh >> 5;
  const int h    = bh & 31;
  const int kvh  = h >> 2;
  const int tid  = threadIdx.x;
  const int wave = tid >> 5;
  const int lane = tid & 31;
  const int q0   = (qblk * 8 + wave) * 16;

  const int cn    = lane & 15;
  const int akb   = (lane >> 4) << 4;
  const int rbase = (lane >> 4) << 3;

  // --- Q fragments: head-indexed RoPE + 1/sqrt(HD), float4 loads ---
  const size_t qbase =
      ((size_t)(b * S_ + q0 + cn)) * (NH_ * HD_) + (size_t)h * HD_;
  auto ropeq4 = [&](int d4) -> v4bf {
    v4f x  = *(const v4f*)&Qp[qbase + d4];
    v4f xr = (d4 < 32) ? -(*(const v4f*)&Qp[qbase + d4 + 32])
                       : *(const v4f*)&Qp[qbase + d4 - 32];
    v4f c  = *(const v4f*)&cosT[h * HD_ + d4];
    v4f sn = *(const v4f*)&sinT[h * HD_ + d4];
    v4f r  = (x * c + xr * sn) * 0.125f;
    v4bf o;
    o[0] = (__bf16)r[0]; o[1] = (__bf16)r[1];
    o[2] = (__bf16)r[2]; o[3] = (__bf16)r[3];
    return o;
  };
  v16bf aq0 = cat16(cat8(ropeq4(akb), ropeq4(akb + 4)),
                    cat8(ropeq4(8 + akb), ropeq4(12 + akb)));
  v16bf aq1 = cat16(cat8(ropeq4(32 + akb), ropeq4(36 + akb)),
                    cat8(ropeq4(40 + akb), ropeq4(44 + akb)));

  float mrow[8], lrow[8];
#pragma unroll
  for (int v = 0; v < 8; ++v) { mrow[v] = -1e30f; lrow[v] = 0.0f; }
  v8f oacc[4] = {v8f{}, v8f{}, v8f{}, v8f{}};

  for (int j0 = 0; j0 < S_; j0 += 32) {
    // ---- batched staging loads ----
    // K: 512 (key,d-quad) items, 2/thread.  V: 256 keypair items, 1/thread.
    int kkey_[2], d4_[2];
    v4f kx[2], kxr[2], kc[2], ks[2];
#pragma unroll
    for (int t = 0; t < 2; ++t) {
      int li   = tid + t * 256;
      int kkey = li >> 4;
      int d4   = (li & 15) * 4;
      kkey_[t] = kkey; d4_[t] = d4;
      size_t kb =
          ((size_t)(b * S_ + j0 + kkey)) * (KVH_ * HD_) + (size_t)kvh * HD_;
      int poff = (d4 < 32) ? 32 : -32;
      kx[t]  = *(const v4f*)&Kp[kb + d4];
      kxr[t] = *(const v4f*)&Kp[kb + d4 + poff];
      kc[t]  = *(const v4f*)&cosT[kvh * HD_ + d4];
      ks[t]  = *(const v4f*)&sinT[kvh * HD_ + d4];
    }
    const int vkey = (tid >> 4) * 2;          // even key of the pair
    const int vd4  = (tid & 15) * 4;
    const size_t vb =
        ((size_t)(b * S_ + j0 + vkey)) * (KVH_ * HD_) + (size_t)kvh * HD_;
    v4f vx0 = *(const v4f*)&Vp[vb + vd4];
    v4f vx1 = *(const v4f*)&Vp[vb + (KVH_ * HD_) + vd4];

    // ---- convert + fragment-swizzled stores (b64 for K, b32 for V) ----
#pragma unroll
    for (int t = 0; t < 2; ++t) {
      int kkey = kkey_[t], d4 = d4_[t];
      float sg = (d4 < 32) ? -1.0f : 1.0f;
      v4f r = kx[t] * kc[t] + kxr[t] * ks[t] * sg;
      v4bf y;
      y[0] = (__bf16)r[0]; y[1] = (__bf16)r[1];
      y[2] = (__bf16)r[2]; y[3] = (__bf16)r[3];
      int kg = ((d4 & 31) >> 4) << 4;          // K-group bit of this d-quad
      *(v4bf*)&Kl[kkey >> 4][d4 >> 5][kg + (kkey & 15)][d4 & 15] = y;
    }
#pragma unroll
    for (int j = 0; j < 4; ++j) {
      int d = vd4 + j;
      v2bf y;                                  // (vkey, vkey+1) adjacent
      y[0] = (__bf16)vx0[j];
      y[1] = (__bf16)vx1[j];
      *(v2bf*)&Vl[d >> 4][(vkey & 16) + (d & 15)][vkey & 15] = y;
    }
    __syncthreads();

    // ---- preload all K and V fragments (16 ds_load_b128) ----
    v16bf bk[2][2], bv[4];
#pragma unroll
    for (int t = 0; t < 2; ++t) {
#pragma unroll
      for (int dh = 0; dh < 2; ++dh) {
        const __bf16* p = &Kl[t][dh][lane][0];
        bk[t][dh] = cat16(*(const v8bf*)p, *(const v8bf*)(p + 8));
      }
    }
#pragma unroll
    for (int c4 = 0; c4 < 4; ++c4) {
      const __bf16* p = &Vl[c4][lane][0];
      bv[c4] = cat16(*(const v8bf*)p, *(const v8bf*)(p + 8));
    }

    // ---- scores: S = Q * K^T (4 WMMAs) ----
    v8f s[2];
#pragma unroll
    for (int t = 0; t < 2; ++t) {
      v8f sc = {};
      sc = wmma_bf16(aq0, bk[t][0], sc);
      sc = wmma_bf16(aq1, bk[t][1], sc);
      s[t] = sc;
    }

    // ---- online softmax over these 32 keys ----
    float alpha[8];
#pragma unroll
    for (int v = 0; v < 8; ++v) {
      float t = fmaxf(s[0][v], s[1][v]);
#pragma unroll
      for (int off = 1; off < 16; off <<= 1)
        t = fmaxf(t, __shfl_xor(t, off, 32));
      float mnew = fmaxf(mrow[v], t);
      alpha[v]   = __expf(mrow[v] - mnew);
      mrow[v]    = mnew;
      float p0 = __expf(s[0][v] - mnew);
      float p1 = __expf(s[1][v] - mnew);
      float ls = p0 + p1;
#pragma unroll
      for (int off = 1; off < 16; off <<= 1)
        ls += __shfl_xor(ls, off, 32);
      lrow[v] = lrow[v] * alpha[v] + ls;
      Plds[wave][rbase + v][cn]      = (__bf16)p0;
      Plds[wave][rbase + v][16 + cn] = (__bf16)p1;
    }
#pragma unroll
    for (int c4 = 0; c4 < 4; ++c4)
#pragma unroll
      for (int v = 0; v < 8; ++v) oacc[c4][v] *= alpha[v];

    lds_wait_wave();  // per-wave LDS region, in-order per wave

    // ---- P as A-fragment, then O += P * V (4 WMMAs) ----
    v16bf ap = cat16(*(const v8bf*)&Plds[wave][cn][akb],
                     *(const v8bf*)&Plds[wave][cn][akb + 8]);
#pragma unroll
    for (int c4 = 0; c4 < 4; ++c4) oacc[c4] = wmma_bf16(ap, bv[c4], oacc[c4]);

    __syncthreads();  // before next staging overwrites Kl/Vl
  }

  // ---- epilogue: normalize and write [B,S,NH*HD] ----
#pragma unroll
  for (int c4 = 0; c4 < 4; ++c4) {
#pragma unroll
    for (int v = 0; v < 8; ++v) {
      int r = rbase + v;
      int d = c4 * 16 + cn;
      Ao[((size_t)(b * S_ + q0 + r)) * (NH_ * HD_) + (size_t)h * HD_ + d] =
          oacc[c4][v] / lrow[v];
    }
  }
}

// ---------------------------------------------------------------------------
extern "C" void kernel_launch(void* const* d_in, const int* in_sizes, int n_in,
                              void* d_out, int out_size, void* d_ws,
                              size_t ws_size, hipStream_t stream) {
  (void)in_sizes; (void)n_in; (void)out_size; (void)ws_size;
  const float* hidden = (const float*)d_in[0];
  const float* Wq     = (const float*)d_in[1];
  const float* Wk     = (const float*)d_in[2];
  const float* Wv     = (const float*)d_in[3];
  const float* Wo     = (const float*)d_in[4];
  const float* cosT   = (const float*)d_in[5];
  const float* sinT   = (const float*)d_in[6];
  float* out = (float*)d_out;

  const size_t M = (size_t)B_ * S_;               // 4096
  float* Qb = (float*)d_ws;                       // [4096, 2048]
  float* Kb = Qb + M * (size_t)(NH_ * HD_);       // [4096, 512]
  float* Vb = Kb + M * (size_t)(KVH_ * HD_);      // [4096, 512]
  float* At = Vb + M * (size_t)(KVH_ * HD_);      // [4096, 2048]

  dim3 blk(256);
  gemm_bf16_wmma_kernel<<<dim3(HID_ / TN, (int)M / TM), blk, 0, stream>>>(
      hidden, Wq, Qb, (int)M, HID_, NH_ * HD_);
  gemm_bf16_wmma_kernel<<<dim3((KVH_ * HD_) / TN, (int)M / TM), blk, 0, stream>>>(
      hidden, Wk, Kb, (int)M, HID_, KVH_ * HD_);
  gemm_bf16_wmma_kernel<<<dim3((KVH_ * HD_) / TN, (int)M / TM), blk, 0, stream>>>(
      hidden, Wv, Vb, (int)M, HID_, KVH_ * HD_);
  flash_attn_kernel<<<dim3(B_ * NH_ * 16), blk, 0, stream>>>(Qb, Kb, Vb, cosT,
                                                             sinT, At);
  gemm_bf16_wmma_kernel<<<dim3(HID_ / TN, (int)M / TM), blk, 0, stream>>>(
      At, Wo, out, (int)M, NH_ * HD_, HID_);
}
